// Qwen3Attention_49864570306561
// MI455X (gfx1250) — matrix operand order repaired and checked by reading, not compiled, not run on hardware
//
#include <hip/hip_runtime.h>
#include <stdint.h>

// ---------------------------------------------------------------------------
// Qwen3 attention block for MI455X (gfx1250, wave32, WMMA).
// Pipeline: cvt(f32->bf16) -> GEMM(qkv) -> rmsnorm+rope split -> flash attn
//           -> GEMM(out proj, f32 result)
// All matmuls use v_wmma_f32_16x16x32_bf16 with f32 accumulation.
// ---------------------------------------------------------------------------

#define HEAD_DIM 128
#define NQH 16
#define NKVH 8
#define BATCH 2
#define SEQ 2048
#define DMODEL 2048
#define EDIM 4096      // qkv projection rows (16+8+8)*128
#define QO_DIM 2048    // 16*128

typedef unsigned short u16;
typedef __attribute__((ext_vector_type(16))) __bf16 v16bf;
typedef __attribute__((ext_vector_type(8)))  float  v8f;

union FragU { v16bf v; uint4 u[2]; };

__device__ __forceinline__ u16 f2bf(float f) {
    unsigned int u = __builtin_bit_cast(unsigned int, f);
    unsigned int r = u + 0x7FFFu + ((u >> 16) & 1u);   // round-to-nearest-even
    return (u16)(r >> 16);
}

__device__ __forceinline__ v8f zero8() {
    v8f z;
    #pragma unroll
    for (int i = 0; i < 8; ++i) z[i] = 0.0f;
    return z;
}

// A operand: 16x32 bf16 (MxK) from row-major [rows][ld] storage.
// ISA layout: lane (0..15)=row M, elements 0..7 -> K = k0 + half*8 + (0..7),
// elements 8..15 -> K = k0 + 16 + half*8 + (0..7).
__device__ __forceinline__ v16bf load_a_frag(const u16* __restrict__ base,
                                             int ld, int m0, int k0, int lane) {
    int half = lane >> 4, ml = lane & 15;
    const u16* p = base + (size_t)(m0 + ml) * ld + k0 + half * 8;
    FragU f;
    f.u[0] = *(const uint4*)(p);
    f.u[1] = *(const uint4*)(p + 16);
    return f.v;
}

// B operand: 32x16 bf16 (KxN) from row-major [Nrows][ld] storage (i.e. B^T,
// K-contiguous). ISA layout: lane (0..15)=column N, elements 0..7 -> K = k0 + half*8 + (0..7),
// elements 8..15 -> K = k0 + 16 + half*8 + (0..7).
__device__ __forceinline__ v16bf load_b_frag(const u16* __restrict__ base,
                                             int ld, int n0, int k0, int lane) {
    int half = lane >> 4, nl = lane & 15;
    const u16* p = base + (size_t)(n0 + nl) * ld + k0 + half * 8;
    FragU f;
    f.u[0] = *(const uint4*)(p);
    f.u[1] = *(const uint4*)(p + 16);
    return f.v;
}

__device__ __forceinline__ v8f wmma_bf16(v16bf a, v16bf b, v8f c) {
    return __builtin_amdgcn_wmma_f32_16x16x32_bf16(
        false, a, false, b, (short)0, c, false, false);
}

// ---------------------------------------------------------------------------
__global__ void cvt_f32_bf16(const float* __restrict__ src,
                             u16* __restrict__ dst, int n) {
    int i = blockIdx.x * blockDim.x + threadIdx.x;
    if (i < n) dst[i] = f2bf(src[i]);
}

// ---------------------------------------------------------------------------
// C[M,N] (f32) = A[M,K] (bf16 row-major) * B[N,K]^T (bf16 row-major)
// 8 waves/block, wave tile 32x64, block tile 64x256.
__global__ __launch_bounds__(256) void gemm_bf16f32(
    const u16* __restrict__ A, const u16* __restrict__ B,
    float* __restrict__ C, int M, int N, int K) {
    int tid = threadIdx.x;
    int wave = tid >> 5, lane = tid & 31;
    int half = lane >> 4, nl = lane & 15;
    int wm = wave >> 2, wn = wave & 3;
    int m0 = blockIdx.y * 64 + wm * 32;
    int n0 = blockIdx.x * 256 + wn * 64;

    v8f acc[2][4];
    #pragma unroll
    for (int i = 0; i < 2; ++i)
        #pragma unroll
        for (int j = 0; j < 4; ++j) acc[i][j] = zero8();

    for (int k0 = 0; k0 < K; k0 += 32) {
        v16bf a0 = load_a_frag(A, K, m0,      k0, lane);
        v16bf a1 = load_a_frag(A, K, m0 + 16, k0, lane);
        v16bf b0 = load_b_frag(B, K, n0,      k0, lane);
        v16bf b1 = load_b_frag(B, K, n0 + 16, k0, lane);
        v16bf b2 = load_b_frag(B, K, n0 + 32, k0, lane);
        v16bf b3 = load_b_frag(B, K, n0 + 48, k0, lane);
        acc[0][0] = wmma_bf16(a0, b0, acc[0][0]);
        acc[0][1] = wmma_bf16(a0, b1, acc[0][1]);
        acc[0][2] = wmma_bf16(a0, b2, acc[0][2]);
        acc[0][3] = wmma_bf16(a0, b3, acc[0][3]);
        acc[1][0] = wmma_bf16(a1, b0, acc[1][0]);
        acc[1][1] = wmma_bf16(a1, b1, acc[1][1]);
        acc[1][2] = wmma_bf16(a1, b2, acc[1][2]);
        acc[1][3] = wmma_bf16(a1, b3, acc[1][3]);
    }
    // C/D layout: lane nl = column, VGPR r -> row r + 8*half.
    #pragma unroll
    for (int i = 0; i < 2; ++i)
        #pragma unroll
        for (int j = 0; j < 4; ++j)
            #pragma unroll
            for (int r = 0; r < 8; ++r)
                C[(size_t)(m0 + 16 * i + r + 8 * half) * N + n0 + 16 * j + nl] =
                    acc[i][j][r];
}

// ---------------------------------------------------------------------------
// Per-head RMS norm + RoPE, split qkv into head-major bf16 buffers.
// One wave per (b,s,head); lane holds d = {lane, lane+32, lane+64, lane+96}.
__global__ __launch_bounds__(256) void norm_rope_split(
    const float* __restrict__ qkv, const float* __restrict__ pcos,
    const float* __restrict__ psin, const float* __restrict__ qw,
    const float* __restrict__ kw, u16* __restrict__ qbf,
    u16* __restrict__ kbf, u16* __restrict__ vt) {
    int tid = threadIdx.x, w = tid >> 5, lane = tid & 31;
    int bs = blockIdx.x;
    int b = bs >> 11, s = bs & (SEQ - 1);
    int head = blockIdx.y * 8 + w;

    const float* src = qkv + (size_t)bs * EDIM + head * HEAD_DIM;
    int d0 = lane, d1 = lane + 32, d2 = lane + 64, d3 = lane + 96;
    float h0 = src[d0], h1 = src[d1], h2 = src[d2], h3 = src[d3];

    if (head < NQH + NKVH) {
        float ss = h0 * h0 + h1 * h1 + h2 * h2 + h3 * h3;
        #pragma unroll
        for (int off = 16; off > 0; off >>= 1) ss += __shfl_xor(ss, off, 32);
        float rinv = rsqrtf(ss * (1.0f / 128.0f) + 1e-6f);
        const float* wn = (head < NQH) ? qw : kw;
        h0 = h0 * rinv * wn[d0];
        h1 = h1 * rinv * wn[d1];
        h2 = h2 * rinv * wn[d2];
        h3 = h3 * rinv * wn[d3];
        const float* cb = pcos + (size_t)s * HEAD_DIM;
        const float* sb = psin + (size_t)s * HEAD_DIM;
        // rotate-half pairs (d, d+64)
        float o0 = h0 * cb[d0] - h2 * sb[d0];
        float o2 = h2 * cb[d2] + h0 * sb[d2];
        float o1 = h1 * cb[d1] - h3 * sb[d1];
        float o3 = h3 * cb[d3] + h1 * sb[d3];
        if (head < NQH) {
            const float sc = 0.08838834764831845f;   // 1/sqrt(128), folded in
            u16* dst = qbf + ((size_t)(b * NQH + head) * SEQ + s) * HEAD_DIM;
            dst[d0] = f2bf(o0 * sc); dst[d1] = f2bf(o1 * sc);
            dst[d2] = f2bf(o2 * sc); dst[d3] = f2bf(o3 * sc);
        } else {
            u16* dst = kbf + ((size_t)(b * NKVH + head - NQH) * SEQ + s) * HEAD_DIM;
            dst[d0] = f2bf(o0); dst[d1] = f2bf(o1);
            dst[d2] = f2bf(o2); dst[d3] = f2bf(o3);
        }
    } else {
        // V: convert and transpose to [b][kvh][d][s] so PV's B operand is
        // K-contiguous.
        int vh = head - NQH - NKVH;
        u16* dst = vt + (size_t)(b * NKVH + vh) * HEAD_DIM * SEQ;
        dst[(size_t)d0 * SEQ + s] = f2bf(h0);
        dst[(size_t)d1 * SEQ + s] = f2bf(h1);
        dst[(size_t)d2 * SEQ + s] = f2bf(h2);
        dst[(size_t)d3 * SEQ + s] = f2bf(h3);
    }
}

// ---------------------------------------------------------------------------
// Causal GQA flash attention. One wave per 16-row Q tile; online softmax with
// f32 running max/sum; P re-layout (C -> A operand) via per-wave LDS tile.
__global__ __launch_bounds__(256) void flash_attn(
    const u16* __restrict__ qbf, const u16* __restrict__ kbf,
    const u16* __restrict__ vt, u16* __restrict__ obf) {
    __shared__ u16 pbuf[8][16][32];

    int tid = threadIdx.x, w = tid >> 5, lane = tid & 31;
    int half = lane >> 4, nl = lane & 15;
    int bh = blockIdx.y;                // b*16 + h
    int b = bh >> 4, h = bh & 15;
    int kh = h >> 1;                    // 2 Q heads per KV head
    int qm = blockIdx.x * 128 + w * 16;

    const u16* qbase = qbf + (size_t)bh * SEQ * HEAD_DIM;
    const u16* kbase = kbf + (size_t)(b * NKVH + kh) * SEQ * HEAD_DIM;
    const u16* vbase = vt  + (size_t)(b * NKVH + kh) * HEAD_DIM * SEQ;

    v16bf qf[4];
    #pragma unroll
    for (int kd = 0; kd < 4; ++kd)
        qf[kd] = load_a_frag(qbase, HEAD_DIM, qm, kd * 32, lane);

    v8f o[8];
    float m[8], l[8];
    #pragma unroll
    for (int c = 0; c < 8; ++c) o[c] = zero8();
    #pragma unroll
    for (int r = 0; r < 8; ++r) { m[r] = -1e30f; l[r] = 0.0f; }

    for (int key0 = 0; key0 <= qm + 15; key0 += 32) {
        // S = Q * K^T for 32 keys (two 16-col subtiles), contraction over 128.
        v8f s0 = zero8(), s1 = zero8();
        #pragma unroll
        for (int kd = 0; kd < 4; ++kd) {
            v16bf kf0 = load_b_frag(kbase, HEAD_DIM, key0,      kd * 32, lane);
            v16bf kf1 = load_b_frag(kbase, HEAD_DIM, key0 + 16, kd * 32, lane);
            s0 = wmma_bf16(qf[kd], kf0, s0);
            s1 = wmma_bf16(qf[kd], kf1, s1);
        }
        if (key0 + 31 > qm) {           // partial (diagonal) tile: causal mask
            #pragma unroll
            for (int r = 0; r < 8; ++r) {
                int row = qm + r + 8 * half;
                if (key0 + nl > row)      s0[r] = -1e30f;
                if (key0 + 16 + nl > row) s1[r] = -1e30f;
            }
        }
        // Online softmax: row reductions across the 16-lane column groups.
        float alpha[8];
        #pragma unroll
        for (int r = 0; r < 8; ++r) {
            float v = fmaxf(s0[r], s1[r]);
            #pragma unroll
            for (int off = 8; off > 0; off >>= 1)
                v = fmaxf(v, __shfl_xor(v, off, 32));
            float mn = fmaxf(m[r], v);
            alpha[r] = __expf(m[r] - mn);
            m[r] = mn;
            float p0 = __expf(s0[r] - mn);
            float p1 = __expf(s1[r] - mn);
            s0[r] = p0; s1[r] = p1;
            float sum = p0 + p1;
            #pragma unroll
            for (int off = 8; off > 0; off >>= 1)
                sum += __shfl_xor(sum, off, 32);
            l[r] = l[r] * alpha[r] + sum;
        }
        #pragma unroll
        for (int c = 0; c < 8; ++c)
            #pragma unroll
            for (int r = 0; r < 8; ++r) o[c][r] *= alpha[r];

        // P: C-layout -> A-layout via per-wave LDS tile (same-wave DS order).
        #pragma unroll
        for (int r = 0; r < 8; ++r) {
            int row = r + 8 * half;
            pbuf[w][row][nl]      = f2bf(s0[r]);
            pbuf[w][row][16 + nl] = f2bf(s1[r]);
        }
        FragU pf;
        pf.u[0] = *(const uint4*)&pbuf[w][nl][half * 8];
        pf.u[1] = *(const uint4*)&pbuf[w][nl][half * 8 + 16];

        // O += P * V  (V^T storage makes B operand K-contiguous)
        #pragma unroll
        for (int c = 0; c < 8; ++c) {
            v16bf vf = load_b_frag(vbase, SEQ, c * 16, key0, lane);
            o[c] = wmma_bf16(pf.v, vf, o[c]);
        }
    }

    float inv[8];
    #pragma unroll
    for (int r = 0; r < 8; ++r) inv[r] = 1.0f / l[r];
    #pragma unroll
    for (int c = 0; c < 8; ++c)
        #pragma unroll
        for (int r = 0; r < 8; ++r) {
            int row = qm + r + 8 * half;
            obf[(size_t)(b * SEQ + row) * QO_DIM + h * HEAD_DIM + c * 16 + nl] =
                f2bf(o[c][r] * inv[r]);
        }
}

// ---------------------------------------------------------------------------
extern "C" void kernel_launch(void* const* d_in, const int* in_sizes, int n_in,
                              void* d_out, int out_size, void* d_ws, size_t ws_size,
                              hipStream_t stream) {
    const float* x    = (const float*)d_in[0];
    const float* pcos = (const float*)d_in[1];
    const float* psin = (const float*)d_in[2];
    const float* wqkv = (const float*)d_in[3];
    const float* wo   = (const float*)d_in[4];
    const float* qw   = (const float*)d_in[5];
    const float* kw   = (const float*)d_in[6];
    float* out = (float*)d_out;

    // workspace layout (bytes), ~152 MB total
    char* ws = (char*)d_ws;
    u16*   x_bf  = (u16*)  (ws + 0ULL);           // 16 MB  x as bf16
    u16*   wq_bf = (u16*)  (ws + 16777216ULL);    // 16 MB  w_qkv bf16
    u16*   wo_bf = (u16*)  (ws + 33554432ULL);    //  8 MB  w_o bf16
    float* qkv   = (float*)(ws + 41943040ULL);    // 64 MB  qkv f32
    u16*   q_bf  = (u16*)  (ws + 109051904ULL);   // 16 MB  q head-major bf16
    u16*   k_bf  = (u16*)  (ws + 125829120ULL);   //  8 MB  k head-major bf16
    u16*   v_t   = (u16*)  (ws + 134217728ULL);   //  8 MB  v transposed bf16
    u16*   a_bf  = (u16*)  (ws + 142606336ULL);   // 16 MB  attn out bf16

    const int nx  = BATCH * SEQ * DMODEL;   // 8388608
    const int nwq = EDIM * DMODEL;          // 8388608
    const int nwo = DMODEL * QO_DIM;        // 4194304

    cvt_f32_bf16<<<(nx  + 255) / 256, 256, 0, stream>>>(x,    x_bf,  nx);
    cvt_f32_bf16<<<(nwq + 255) / 256, 256, 0, stream>>>(wqkv, wq_bf, nwq);
    cvt_f32_bf16<<<(nwo + 255) / 256, 256, 0, stream>>>(wo,   wo_bf, nwo);

    // qkv[bs][e] = x[bs][:] . w_qkv[e][:]
    dim3 g1(EDIM / 256, (BATCH * SEQ) / 64);
    gemm_bf16f32<<<g1, 256, 0, stream>>>(x_bf, wq_bf, qkv,
                                         BATCH * SEQ, EDIM, DMODEL);

    dim3 g2(BATCH * SEQ, 4);
    norm_rope_split<<<g2, 256, 0, stream>>>(qkv, pcos, psin, qw, kw,
                                            q_bf, k_bf, v_t);

    dim3 g3(SEQ / 128, BATCH * NQH);
    flash_attn<<<g3, 256, 0, stream>>>(q_bf, k_bf, v_t, a_bf);

    // out[bs][d] = attn[bs][:] . w_o[d][:]
    dim3 g4(DMODEL / 256, (BATCH * SEQ) / 64);
    gemm_bf16f32<<<g4, 256, 0, stream>>>(a_bf, wo_bf, out,
                                         BATCH * SEQ, DMODEL, QO_DIM);
}
